// MHSA_41025527611405
// MI455X (gfx1250) — matrix operand order repaired and checked by reading, not compiled, not run on hardware
//
#include <hip/hip_runtime.h>

typedef __attribute__((ext_vector_type(16))) _Float16 v16h;
typedef __attribute__((ext_vector_type(8)))  float    v8f;
typedef __attribute__((ext_vector_type(4)))  _Float16 h4;
typedef __attribute__((ext_vector_type(2)))  _Float16 h2;

#define BATCH 256
#define NTOK  196
#define CDIM  784
#define HEADS 4
#define HDIM  196
#define MROWS (BATCH * NTOK)          // 50176
#define PERH  (NTOK * HDIM)           // 38416
#define TS    232                      // LDS stride (halves) for qT/k/vT tiles
#define PS    200                      // LDS stride (halves) for P staging
#define ATTN_SMEM_HALVES (3 * NTOK * TS + 8 * 16 * PS)   // 162016
#define ATTN_SMEM_BYTES  (ATTN_SMEM_HALVES * 2)          // 324032 <= 320KB WGP LDS

// ---------------------------------------------------------------------------
// WMMA helpers — layouts per CDNA5 ISA 7.12.2 (wave32)
// ---------------------------------------------------------------------------
__device__ __forceinline__ v8f wmma_f16(v16h a, v16h b, v8f c) {
  return __builtin_amdgcn_wmma_f32_16x16x32_f16(false, a, false, b, (short)0, c,
                                                false, false);
}

// A-frag (16x32 f16): lane holds row (lane&15); element e -> K = e+(e>=8?8:0)+8*(lane>>4)
// Row-major storage (row, k): contiguous pairs.
__device__ __forceinline__ v16h ld_a(const _Float16* s, int stride, int row_base,
                                     int k_base, int row_lim, int k_lim) {
  const int lane = threadIdx.x & 31;
  const int r = row_base + (lane & 15);
  const int lh = lane >> 4;
  const bool rok = r < row_lim;
  v16h a;
  h2 z2 = {(_Float16)0.f, (_Float16)0.f};
#pragma unroll
  for (int p = 0; p < 8; ++p) {
    int e = 2 * p;
    int k = k_base + e + (e >= 8 ? 8 : 0) + 8 * lh;
    h2 c = (rok && (k + 1) < k_lim) ? *(const h2*)(s + (size_t)r * stride + k) : z2;
    a[e] = c[0];
    a[e + 1] = c[1];
  }
  return a;
}

__device__ __forceinline__ v16h ld_a_np(const _Float16* s, int stride, int row_base,
                                        int k_base) {
  const int lane = threadIdx.x & 31;
  const int r = row_base + (lane & 15);
  const int lh = lane >> 4;
  v16h a;
#pragma unroll
  for (int p = 0; p < 8; ++p) {
    int e = 2 * p;
    int k = k_base + e + (e >= 8 ? 8 : 0) + 8 * lh;
    h2 c = *(const h2*)(s + (size_t)r * stride + k);
    a[e] = c[0];
    a[e + 1] = c[1];
  }
  return a;
}

// A-frag from K-major storage (k, row): strided (used only 7x per strip).
__device__ __forceinline__ v16h ld_a_cols(const _Float16* s, int stride, int row_base,
                                          int k_base, int row_lim, int k_lim) {
  const int lane = threadIdx.x & 31;
  const int r = row_base + (lane & 15);
  const int lh = lane >> 4;
  const bool rok = r < row_lim;
  v16h a;
#pragma unroll
  for (int e = 0; e < 16; ++e) {
    int k = k_base + e + (e >= 8 ? 8 : 0) + 8 * lh;
    a[e] = (rok && k < k_lim) ? s[(size_t)k * stride + r] : (_Float16)0.f;
  }
  return a;
}

// B-frag (32x16 f16): lane holds col (lane&15); element e -> K = e+(e>=8?8:0)+8*(lane>>4).
// Storage row index == B column, contraction contiguous in row -> b128-friendly.
__device__ __forceinline__ v16h ld_b_rows(const _Float16* s, int stride, int col_base,
                                          int k_base, int col_lim, int k_lim) {
  const int lane = threadIdx.x & 31;
  const int c = col_base + (lane & 15);
  const int lh = lane >> 4;
  const bool cok = c < col_lim;
  v16h b;
  h4 z4 = {(_Float16)0.f, (_Float16)0.f, (_Float16)0.f, (_Float16)0.f};
#pragma unroll
  for (int j = 0; j < 4; ++j) {
    int k = k_base + 8 * lh + 4 * j + (j >= 2 ? 8 : 0);
    h4 v = (cok && (k + 3) < k_lim) ? *(const h4*)(s + (size_t)c * stride + k) : z4;
#pragma unroll
    for (int w = 0; w < 4; ++w) b[4 * j + w] = v[w];
  }
  return b;
}

__device__ __forceinline__ v16h ld_b_rows_np(const _Float16* s, int stride,
                                             int col_base, int k_base) {
  const int lane = threadIdx.x & 31;
  const int c = col_base + (lane & 15);
  const int lh = lane >> 4;
  v16h b;
#pragma unroll
  for (int j = 0; j < 4; ++j) {
    int k = k_base + 8 * lh + 4 * j + (j >= 2 ? 8 : 0);
    h4 v = *(const h4*)(s + (size_t)c * stride + k);
#pragma unroll
    for (int w = 0; w < 4; ++w) b[4 * j + w] = v[w];
  }
  return b;
}

// CDNA5 async global->LDS copy (8 bytes/lane), tracked on ASYNCcnt.
// LDS address = low 32 bits of the flat address (LDS aperture truncation).
__device__ __forceinline__ void async_copy_b64(const _Float16* g, _Float16* l) {
  unsigned loff = (unsigned)(unsigned long long)(const void*)l;
  asm volatile("global_load_async_to_lds_b64 %0, %1, off"
               :: "v"(loff), "v"(g) : "memory");
}

// ---------------------------------------------------------------------------
// Stage 0: conversions
// ---------------------------------------------------------------------------
__global__ void f32_to_f16_kernel(const float* __restrict__ src,
                                  _Float16* __restrict__ dst, int n) {
  int i = blockIdx.x * 256 + threadIdx.x;
  if (i < n) dst[i] = (_Float16)src[i];
}

__global__ void build_cp_kernel(const float* __restrict__ rel_h,
                                const float* __restrict__ rel_w,
                                _Float16* __restrict__ cp) {
  int i = blockIdx.x * 256 + threadIdx.x;
  if (i >= HEADS * PERH) return;
  int h = i / PERH;
  int rem = i % PERH;
  int p = rem / HDIM;    // token position (HW)
  int kd = rem % HDIM;   // channel d (contraction in pos matmul)
  float v = rel_h[((size_t)h * HDIM + kd) * 14 + (p / 14)] +
            rel_w[((size_t)h * HDIM + kd) * 14 + (p % 14)];
  cp[i] = (_Float16)v;   // cp[h][p][kd]
}

// ---------------------------------------------------------------------------
// Stage 1/3: GEMM  out = A(MxK) * W(NxK)^T + bias.
// mode 0: f16 head-split store; z==0 (q) and z==2 (v) stored transposed
//         (b,h,d,n); z==1 (k) stored (b,h,n,d) and pre-scaled by 1/sqrt(d).
// mode 1: f32 row-major store.
// ---------------------------------------------------------------------------
__global__ __launch_bounds__(256) void gemm_wmma_kernel(
    const _Float16* __restrict__ A, const _Float16* __restrict__ W0,
    const _Float16* __restrict__ W1, const _Float16* __restrict__ W2,
    const float* __restrict__ b0, const float* __restrict__ b1,
    const float* __restrict__ b2, void* o0, void* o1, void* o2, int M, int N,
    int K, int mode) {
  __shared__ _Float16 sA[2][128 * 40];
  __shared__ _Float16 sB[2][128 * 40];

  const _Float16* W = (blockIdx.z == 0) ? W0 : (blockIdx.z == 1 ? W1 : W2);
  const float* bias = (blockIdx.z == 0) ? b0 : (blockIdx.z == 1 ? b1 : b2);
  void* outp = (blockIdx.z == 0) ? o0 : (blockIdx.z == 1 ? o1 : o2);

  const int bm0 = blockIdx.y * 128;
  const int bn0 = blockIdx.x * 128;

  auto load_stage = [&](int kt, int buf) {
    int k0 = kt * 32;
    h4 z4 = {(_Float16)0.f, (_Float16)0.f, (_Float16)0.f, (_Float16)0.f};
#pragma unroll
    for (int i = 0; i < 4; ++i) {
      int c = threadIdx.x + i * 256;       // 1024 chunks of 4 halves
      int row = c >> 3;
      int col = (c & 7) << 2;
      int kg = k0 + col;
      h4 va = (kg < K) ? *(const h4*)(A + (size_t)(bm0 + row) * K + kg) : z4;
      *(h4*)(&sA[buf][row * 40 + col]) = va;
      int nn = bn0 + row;
      h4 vb = (kg < K && nn < N) ? *(const h4*)(W + (size_t)nn * K + kg) : z4;
      *(h4*)(&sB[buf][row * 40 + col]) = vb;
    }
  };

  const int wave = threadIdx.x >> 5;
  const int wm = wave >> 1;      // 0..3 -> 32-row strip
  const int wn = wave & 1;       // 0..1 -> 64-col strip
  v8f acc[2][4];
  v8f zero = {0.f, 0.f, 0.f, 0.f, 0.f, 0.f, 0.f, 0.f};
#pragma unroll
  for (int i = 0; i < 2; ++i)
#pragma unroll
    for (int j = 0; j < 4; ++j) acc[i][j] = zero;

  const int KT = (K + 31) / 32;
  load_stage(0, 0);
  for (int kt = 0; kt < KT; ++kt) {
    __syncthreads();
    if (kt + 1 < KT) load_stage(kt + 1, (kt + 1) & 1);
    const _Float16* cA = sA[kt & 1];
    const _Float16* cB = sB[kt & 1];
    v16h a0 = ld_a_np(cA, 40, wm * 32, 0);
    v16h a1 = ld_a_np(cA, 40, wm * 32 + 16, 0);
#pragma unroll
    for (int j = 0; j < 4; ++j) {
      v16h bb = ld_b_rows_np(cB, 40, wn * 64 + j * 16, 0);
      acc[0][j] = wmma_f16(a0, bb, acc[0][j]);
      acc[1][j] = wmma_f16(a1, bb, acc[1][j]);
    }
  }

  const int lane = threadIdx.x & 31;
  const int nlo = lane & 15;
  const int lh = lane >> 4;
  const float kscale = (mode == 0 && blockIdx.z == 1) ? (1.0f / 14.0f) : 1.0f;
  const bool transposed = (mode == 0) && (blockIdx.z != 1);
#pragma unroll
  for (int i = 0; i < 2; ++i)
#pragma unroll
    for (int j = 0; j < 4; ++j)
#pragma unroll
      for (int r = 0; r < 8; ++r) {
        int row = bm0 + wm * 32 + i * 16 + r + 8 * lh;
        int col = bn0 + wn * 64 + j * 16 + nlo;
        if (col < N) {
          float val = (acc[i][j][r] + bias[col]) * kscale;
          if (mode == 0) {
            int bi = row / NTOK, n = row % NTOK;
            int hh = col / HDIM, dd = col % HDIM;
            size_t base = ((size_t)bi * HEADS + hh);
            size_t idx = transposed ? (base * HDIM + dd) * NTOK + n
                                    : (base * NTOK + n) * HDIM + dd;
            ((_Float16*)outp)[idx] = (_Float16)val;
          } else {
            ((float*)outp)[(size_t)row * N + col] = val;
          }
        }
      }
}

// ---------------------------------------------------------------------------
// Stage 2: attention per (b,h).  logits = q*(k/14)^T + cp*q, softmax, *v.
// qT/vT are (d, token) in global; k is (token, d) pre-scaled by 1/14.
// ---------------------------------------------------------------------------
__global__ __launch_bounds__(256) void attn_wmma_kernel(
    const _Float16* __restrict__ qth, const _Float16* __restrict__ kh,
    const _Float16* __restrict__ vth, const _Float16* __restrict__ cph,
    _Float16* __restrict__ oh) {
  extern __shared__ _Float16 smem[];
  _Float16* sQT = smem;                  // (d, token) stride TS
  _Float16* sK  = smem + NTOK * TS;      // (token, d) stride TS, pre-scaled
  _Float16* sVT = smem + 2 * NTOK * TS;  // (d, token) stride TS
  _Float16* sP  = smem + 3 * NTOK * TS;  // 8 waves x 16 x PS

  const int bh = blockIdx.x;
  const int b = bh >> 2;
  const int h = bh & 3;
  const _Float16* gqt = qth + (size_t)bh * PERH;
  const _Float16* gk  = kh  + (size_t)bh * PERH;
  const _Float16* gvt = vth + (size_t)bh * PERH;
  const _Float16* cp  = cph + (size_t)h * PERH;

  // Async tile copies (global -> LDS), 8B/lane, drained on ASYNCcnt.
  for (int c = threadIdx.x; c < NTOK * 49; c += 256) {
    int row = c / 49;
    int col = (c % 49) * 4;
    async_copy_b64(gqt + row * HDIM + col, sQT + row * TS + col);
    async_copy_b64(gk  + row * HDIM + col, sK  + row * TS + col);
    async_copy_b64(gvt + row * HDIM + col, sVT + row * TS + col);
  }
  asm volatile("s_wait_asynccnt 0" ::: "memory");
  __syncthreads();

  const int wave = threadIdx.x >> 5;
  const int lane = threadIdx.x & 31;
  const int nlo = lane & 15;
  const int lh = lane >> 4;
  _Float16* myP = sP + wave * (16 * PS);

  for (int mt = wave; mt < 13; mt += 8) {
    const int rowbase = mt * 16;
    v8f acc[13];
    v8f zero = {0.f, 0.f, 0.f, 0.f, 0.f, 0.f, 0.f, 0.f};
#pragma unroll
    for (int nt = 0; nt < 13; ++nt) acc[nt] = zero;

    __builtin_prefetch(cp + (size_t)rowbase * HDIM, 0, 1);

    v16h afr[7];
    // pos = cp(rows p) x q : B element (k=token, col=d) = sQT[d*TS + token]
#pragma unroll
    for (int kt = 0; kt < 7; ++kt)
      afr[kt] = ld_a(cp, HDIM, rowbase, kt * 32, NTOK, NTOK);
#pragma unroll
    for (int nt = 0; nt < 13; ++nt)
#pragma unroll
      for (int kt = 0; kt < 7; ++kt) {
        v16h bfr = ld_b_rows(sQT, TS, nt * 16, kt * 32, HDIM, NTOK);
        acc[nt] = wmma_f16(afr[kt], bfr, acc[nt]);
      }

    // sim = q(rows n) x k^T : A element (row n, k=d) = sQT[d*TS + n]
#pragma unroll
    for (int kt = 0; kt < 7; ++kt)
      afr[kt] = ld_a_cols(sQT, TS, rowbase, kt * 32, NTOK, HDIM);
#pragma unroll
    for (int nt = 0; nt < 13; ++nt)
#pragma unroll
      for (int kt = 0; kt < 7; ++kt) {
        v16h bfr = ld_b_rows(sK, TS, nt * 16, kt * 32, NTOK, HDIM);
        acc[nt] = wmma_f16(afr[kt], bfr, acc[nt]);
      }

    // Mask padded attention columns (196..207) before softmax.
    if (nlo >= 4) {
#pragma unroll
      for (int r = 0; r < 8; ++r) acc[12][r] = -1e30f;
    }

    // Row softmax: reduce across 13 frags then 16 lanes (halves independent).
    float rinv[8];
#pragma unroll
    for (int r = 0; r < 8; ++r) {
      float m = acc[0][r];
#pragma unroll
      for (int nt = 1; nt < 13; ++nt) m = fmaxf(m, acc[nt][r]);
#pragma unroll
      for (int s = 1; s < 16; s <<= 1) m = fmaxf(m, __shfl_xor(m, s, 16));
      float sum = 0.f;
#pragma unroll
      for (int nt = 0; nt < 13; ++nt) {
        float e = __expf(acc[nt][r] - m);
        acc[nt][r] = e;
        sum += e;
      }
#pragma unroll
      for (int s = 1; s < 16; s <<= 1) sum += __shfl_xor(sum, s, 16);
      rinv[r] = 1.f / sum;
    }

    // Stage P (C-layout -> row-major) through per-wave LDS buffer.
#pragma unroll
    for (int nt = 0; nt < 13; ++nt)
#pragma unroll
      for (int r = 0; r < 8; ++r) {
        int col = nt * 16 + nlo;
        if (col < PS)
          myP[(r + 8 * lh) * PS + col] = (_Float16)(acc[nt][r] * rinv[r]);
      }
    asm volatile("s_wait_dscnt 0" ::: "memory");  // in-wave LDS RAW across lanes

    // out = P(16xN) x v : B element (k=token, col=d) = sVT[d*TS + token]
#pragma unroll
    for (int kt = 0; kt < 7; ++kt) afr[kt] = ld_a(myP, PS, 0, kt * 32, 16, NTOK);
#pragma unroll
    for (int nt = 0; nt < 13; ++nt) acc[nt] = zero;
#pragma unroll
    for (int nt = 0; nt < 13; ++nt)
#pragma unroll
      for (int kt = 0; kt < 7; ++kt) {
        v16h bfr = ld_b_rows(sVT, TS, nt * 16, kt * 32, HDIM, NTOK);
        acc[nt] = wmma_f16(afr[kt], bfr, acc[nt]);
      }

    // Store strip to (b, n, h*196 + d) f16.
#pragma unroll
    for (int nt = 0; nt < 13; ++nt)
#pragma unroll
      for (int r = 0; r < 8; ++r) {
        int n = rowbase + r + 8 * lh;
        int dcol = nt * 16 + nlo;
        if (n < NTOK && dcol < HDIM)
          oh[((size_t)b * NTOK + n) * CDIM + h * HDIM + dcol] = (_Float16)acc[nt][r];
      }
  }
}

// ---------------------------------------------------------------------------
// Host launcher
// ---------------------------------------------------------------------------
static inline size_t al256(size_t x) { return (x + 255) & ~(size_t)255; }

extern "C" void kernel_launch(void* const* d_in, const int* in_sizes, int n_in,
                              void* d_out, int out_size, void* d_ws, size_t ws_size,
                              hipStream_t stream) {
  const float* x   = (const float*)d_in[0];
  const float* Wq  = (const float*)d_in[1];
  const float* bq  = (const float*)d_in[2];
  const float* Wk  = (const float*)d_in[3];
  const float* bk  = (const float*)d_in[4];
  const float* Wv  = (const float*)d_in[5];
  const float* bv  = (const float*)d_in[6];
  const float* Wo  = (const float*)d_in[7];
  const float* bo  = (const float*)d_in[8];
  const float* rlh = (const float*)d_in[9];
  const float* rlw = (const float*)d_in[10];

  const size_t XH_BYTES = (size_t)MROWS * CDIM * 2;       // 78,675,968 (aliased by OH)
  const size_t W_BYTES  = (size_t)CDIM * CDIM * 2;        // 1,229,312
  const size_t CP_BYTES = (size_t)HEADS * PERH * 2;       // 307,328
  const size_t QH_BYTES = XH_BYTES;

  char* ws = (char*)d_ws;
  size_t off = 0;
  _Float16* xh  = (_Float16*)(ws + off); off = al256(off + XH_BYTES);
  _Float16* wqh = (_Float16*)(ws + off); off = al256(off + W_BYTES);
  _Float16* wkh = (_Float16*)(ws + off); off = al256(off + W_BYTES);
  _Float16* wvh = (_Float16*)(ws + off); off = al256(off + W_BYTES);
  _Float16* woh = (_Float16*)(ws + off); off = al256(off + W_BYTES);
  _Float16* cph = (_Float16*)(ws + off); off = al256(off + CP_BYTES);
  _Float16* qtp = (_Float16*)(ws + off); off = al256(off + QH_BYTES);  // q (b,h,d,n)
  _Float16* khp = (_Float16*)(ws + off); off = al256(off + QH_BYTES);  // k (b,h,n,d)/14
  _Float16* vtp = (_Float16*)(ws + off); off = al256(off + QH_BYTES);  // v (b,h,d,n)
  _Float16* ohp = xh;  // xh is dead after the QKV GEMM -> reuse for attention out

  const int NX = MROWS * CDIM;       // 39,337,984
  const int NW = CDIM * CDIM;        // 614,656
  f32_to_f16_kernel<<<(NX + 255) / 256, 256, 0, stream>>>(x, xh, NX);
  f32_to_f16_kernel<<<(NW + 255) / 256, 256, 0, stream>>>(Wq, wqh, NW);
  f32_to_f16_kernel<<<(NW + 255) / 256, 256, 0, stream>>>(Wk, wkh, NW);
  f32_to_f16_kernel<<<(NW + 255) / 256, 256, 0, stream>>>(Wv, wvh, NW);
  f32_to_f16_kernel<<<(NW + 255) / 256, 256, 0, stream>>>(Wo, woh, NW);
  build_cp_kernel<<<(HEADS * PERH + 255) / 256, 256, 0, stream>>>(rlh, rlw, cph);

  // QKV projections (z selects q/k/v; q,v stored transposed, k pre-scaled).
  gemm_wmma_kernel<<<dim3(7, MROWS / 128, 3), 256, 0, stream>>>(
      xh, wqh, wkh, wvh, bq, bk, bv, (void*)qtp, (void*)khp, (void*)vtp, MROWS,
      CDIM, CDIM, 0);

  // Attention: one workgroup per (b,h); 324 KB dynamic LDS (CDNA5 WGP: 320KB).
  (void)hipFuncSetAttribute((const void*)attn_wmma_kernel,
                            hipFuncAttributeMaxDynamicSharedMemorySize,
                            ATTN_SMEM_BYTES);
  attn_wmma_kernel<<<BATCH * HEADS, 256, ATTN_SMEM_BYTES, stream>>>(qtp, khp, vtp,
                                                                    cph, ohp);

  // Output projection -> f32 d_out.
  gemm_wmma_kernel<<<dim3(7, MROWS / 128, 1), 256, 0, stream>>>(
      ohp, woh, woh, woh, bo, bo, bo, d_out, d_out, d_out, MROWS, CDIM, CDIM, 1);
}
